// BaseLayerWithLoRA_66718021976180
// MI455X (gfx1250) — matrix operand order repaired and checked by reading, not compiled, not run on hardware
//
#include <hip/hip_runtime.h>

// ---------------- problem constants ----------------
constexpr int Nn  = 32768;
constexpr int Dk  = 2048;      // D_IN  (K of base GEMM)
constexpr int Do  = 2048;      // D_OUT
constexpr int Ll  = 8;
constexpr int Rr  = 16;
constexpr int LR  = Ll * Rr;   // 128 extra K columns
constexpr int Ke  = Dk + LR;   // 2176 extended K
constexpr float SCALE = 2.0f;  // 32/16

// ---------------- tiling ----------------
constexpr int BM = 128, BN = 128, BK = 32;   // block tile; 8 waves: 4(M) x 2(N)

typedef __attribute__((ext_vector_type(16))) __bf16 v16bf;
typedef __attribute__((ext_vector_type(8)))  float  v8f;

struct Pair16 { uint4 lo, hi; };   // 32 bytes -> v16bf

__device__ inline unsigned short f2bf(float f) {
  unsigned int u = __builtin_bit_cast(unsigned int, f);
  u += 0x7FFFu + ((u >> 16) & 1u);          // round-to-nearest-even
  return (unsigned short)(u >> 16);
}

// A fragment (16x32, 16-bit): lanes<16 hold K[0..8)+K[16..24), lanes>=16 K[8..16)+K[24..32)
__device__ inline v16bf load_frag_a(const unsigned short* base, int lane) {
  const unsigned short* p = base + (lane & 15) * BK + ((lane >> 4) ? 8 : 0);
  Pair16 t;
  t.lo = *(const uint4*)(p);
  t.hi = *(const uint4*)(p + 16);
  return __builtin_bit_cast(v16bf, t);
}
// B fragment (32x16, 16-bit): lane holds one N-column = one W-row; lanes<16 K[0..8)+K[16..24), lanes>=16 K[8..16)+K[24..32)
__device__ inline v16bf load_frag_b(const unsigned short* base, int lane) {
  const unsigned short* p = base + (lane & 15) * BK + ((lane >> 4) ? 8 : 0);
  Pair16 t;
  t.lo = *(const uint4*)(p);
  t.hi = *(const uint4*)(p + 16);
  return __builtin_bit_cast(v16bf, t);
}

// CDNA5 async copy: copies global saddr + vaddr into LDS at lds_addr, 16 bytes, ASYNCcnt-tracked
__device__ inline void async_copy_b128(unsigned lds_addr, unsigned voff, const void* sbase) {
  asm volatile("global_load_async_to_lds_b128 %0, %1, %2"
               :: "v"(lds_addr), "v"(voff), "s"(sbase)
               : "memory");
}
__device__ inline void wait_async0() {
  asm volatile("s_wait_asynccnt 0x0" ::: "memory");
}

// ---------------- WMMA GEMM: C[M,N] = A[M,K] * B[N,K]^T ----------------
// MODE 0: out[row*Do+col]  = acc + bias[col]                (fp32)
// MODE 1: Xb[row*Ke+Dk+col] = bf16(mask * SCALE * acc)      (masked LoRA U)
template<int MODE, int LDA, int LDB, int KTOT>
__global__ __launch_bounds__(256)
void gemm_wmma(const unsigned short* __restrict__ Amat,
               const unsigned short* __restrict__ Bmat,
               float* __restrict__ outF, const float* __restrict__ bias,
               unsigned short* __restrict__ outU,
               const int* __restrict__ lidx) {
  const int tid  = threadIdx.x;
  const int lane = tid & 31;
  const int wave = tid >> 5;
  const int wm   = wave & 3;            // 0..3 -> 32-row band
  const int wn   = wave >> 2;           // 0..1 -> 64-col band
  const int bm   = blockIdx.x * BM;
  const int bn   = blockIdx.y * BN;

  __shared__ __align__(16) unsigned short As[2][BM * BK];
  __shared__ __align__(16) unsigned short Bs[2][BN * BK];

  v8f acc[2][4];
#pragma unroll
  for (int mi = 0; mi < 2; ++mi)
#pragma unroll
    for (int ni = 0; ni < 4; ++ni)
      acc[mi][ni] = (v8f){0.f,0.f,0.f,0.f,0.f,0.f,0.f,0.f};

  // per-thread copy chunks: tile is (128 rows x 32 k) bf16 = 512 x 16B; 256 threads x 2 chunks
  const int c0   = tid;                 // chunk 0: rows 0..63
  const int row0 = c0 >> 2, kc0 = (c0 & 3) * 8;
  const int row1 = row0 + 64;           // chunk 1: rows 64..127, same kc
  const unsigned voffA0 = (unsigned)((row0 * LDA + kc0) * 2);
  const unsigned voffA1 = (unsigned)((row1 * LDA + kc0) * 2);
  const unsigned voffB0 = (unsigned)((row0 * LDB + kc0) * 2);
  const unsigned voffB1 = (unsigned)((row1 * LDB + kc0) * 2);
  const unsigned ldsA = (unsigned)(uintptr_t)(&As[0][0]) + (unsigned)((row0 * BK + kc0) * 2);
  const unsigned ldsB = (unsigned)(uintptr_t)(&Bs[0][0]) + (unsigned)((row0 * BK + kc0) * 2);
  constexpr unsigned BUFB  = (unsigned)(BM * BK * 2);   // 8192 bytes per buffer
  constexpr unsigned HALF  = (unsigned)(64 * BK * 2);   // 4096 bytes: rows 64..127

  const char* Abase = (const char*)Amat + (size_t)bm * LDA * 2;
  const char* Bbase = (const char*)Bmat + (size_t)bn * LDB * 2;

  auto issue_tile = [&](int kt, int buf) {
    const char* sa = Abase + (size_t)kt * (BK * 2);
    const char* sb = Bbase + (size_t)kt * (BK * 2);
    const unsigned bo = (unsigned)buf * BUFB;
    async_copy_b128(ldsA + bo,        voffA0, sa);
    async_copy_b128(ldsA + bo + HALF, voffA1, sa);
    async_copy_b128(ldsB + bo,        voffB0, sb);
    async_copy_b128(ldsB + bo + HALF, voffB1, sb);
  };

  constexpr int KT = KTOT / BK;
  issue_tile(0, 0);
  wait_async0();
  __syncthreads();

#pragma unroll 2
  for (int kt = 0; kt < KT; ++kt) {
    const int cur = kt & 1;
    if (kt + 1 < KT) issue_tile(kt + 1, cur ^ 1);   // async prefetch into other buffer

    const unsigned short* Abp = &As[cur][(wm * 32) * BK];
    const unsigned short* Bbp = &Bs[cur][(wn * 64) * BK];
    v16bf af[2], bf[4];
#pragma unroll
    for (int mi = 0; mi < 2; ++mi) af[mi] = load_frag_a(Abp + mi * 16 * BK, lane);
#pragma unroll
    for (int ni = 0; ni < 4; ++ni) bf[ni] = load_frag_b(Bbp + ni * 16 * BK, lane);
#pragma unroll
    for (int mi = 0; mi < 2; ++mi)
#pragma unroll
      for (int ni = 0; ni < 4; ++ni)
        acc[mi][ni] = __builtin_amdgcn_wmma_f32_16x16x32_bf16(
            false, af[mi], false, bf[ni], (short)0, acc[mi][ni], false, false);

    if (kt + 1 < KT) wait_async0();     // prefetch done before everyone flips buffers
    __syncthreads();
  }

  // ---------------- epilogue ----------------
  const int rbase = bm + wm * 32;
  const int cbase = bn + wn * 64;
  const int cl = lane & 15;
  const int rh = (lane >> 4) * 8;       // C layout: VGPR r -> M = r (+8 for upper half-wave)

  if (MODE == 0) {
#pragma unroll
    for (int mi = 0; mi < 2; ++mi)
#pragma unroll
      for (int ni = 0; ni < 4; ++ni) {
        const int col = cbase + ni * 16 + cl;
        const float bb = bias[col];
#pragma unroll
        for (int r = 0; r < 8; ++r) {
          const int row = rbase + mi * 16 + rh + r;
          outF[(size_t)row * Do + col] = acc[mi][ni][r] + bb;
        }
      }
  } else {
#pragma unroll
    for (int mi = 0; mi < 2; ++mi)
#pragma unroll
      for (int r = 0; r < 8; ++r) {
        const int row = rbase + mi * 16 + rh + r;
        const int idxn = lidx[row];
#pragma unroll
        for (int ni = 0; ni < 4; ++ni) {
          const int col = cbase + ni * 16 + cl;        // 0..127
          const float v = (idxn == (col >> 4)) ? acc[mi][ni][r] * SCALE : 0.f;
          outU[(size_t)row * Ke + Dk + col] = f2bf(v);
        }
      }
  }
}

// ---------------- staging kernels ----------------
__global__ void k_build_wb(const float* __restrict__ W, const float* __restrict__ Ball,
                           unsigned short* __restrict__ Wb) {
  size_t i = (size_t)blockIdx.x * 256 + threadIdx.x;
  if (i >= (size_t)Do * Ke) return;
  int o = (int)(i / Ke), k = (int)(i % Ke);
  float v;
  if (k < Dk) v = W[(size_t)o * Dk + k];
  else { int c = k - Dk; v = Ball[((size_t)(c >> 4) * Do + o) * Rr + (c & 15)]; }
  Wb[i] = f2bf(v);
}

__global__ void k_build_ab(const float* __restrict__ Aall, unsigned short* __restrict__ Ab) {
  size_t i = (size_t)blockIdx.x * 256 + threadIdx.x;
  if (i >= (size_t)LR * Dk) return;
  Ab[i] = f2bf(Aall[i]);
}

__global__ void k_convert_x(const float* __restrict__ x, unsigned short* __restrict__ Xb) {
  size_t i = (size_t)blockIdx.x * 256 + threadIdx.x;   // N*Dk/8 threads
  size_t n = i / (Dk / 8);
  int kk = (int)(i % (Dk / 8)) * 8;
  const float4* p = (const float4*)(x + n * Dk + kk);
  float4 a = p[0], b = p[1];
  uint4 o;
  o.x = (unsigned)f2bf(a.x) | ((unsigned)f2bf(a.y) << 16);
  o.y = (unsigned)f2bf(a.z) | ((unsigned)f2bf(a.w) << 16);
  o.z = (unsigned)f2bf(b.x) | ((unsigned)f2bf(b.y) << 16);
  o.w = (unsigned)f2bf(b.z) | ((unsigned)f2bf(b.w) << 16);
  *(uint4*)(Xb + n * Ke + kk) = o;
}

// ---------------- no-workspace fallback (insurance only) ----------------
__global__ void k_fb_base(const float* __restrict__ x, const float* __restrict__ W,
                          const float* __restrict__ b, float* __restrict__ out) {
  size_t i = (size_t)blockIdx.x * 256 + threadIdx.x;
  if (i >= (size_t)Nn * Do) return;
  int n = (int)(i / Do), o = (int)(i % Do);
  float s = b[o];
  for (int k = 0; k < Dk; ++k) s += x[(size_t)n * Dk + k] * W[(size_t)o * Dk + k];
  out[i] = s;
}
__global__ void k_fb_delta(const float* __restrict__ x, const float* __restrict__ Aall,
                           const float* __restrict__ Ball, const int* __restrict__ lidx,
                           float* __restrict__ out) {
  int n = blockIdx.x, tid = threadIdx.x;
  int idx = lidx[n];
  if (idx < 0) return;
  __shared__ float red[256];
  __shared__ float u[Rr];
  for (int r = 0; r < Rr; ++r) {
    float s = 0.f;
    for (int k = tid; k < Dk; k += 256)
      s += x[(size_t)n * Dk + k] * Aall[((size_t)idx * Rr + r) * Dk + k];
    red[tid] = s; __syncthreads();
    for (int st = 128; st > 0; st >>= 1) { if (tid < st) red[tid] += red[tid + st]; __syncthreads(); }
    if (tid == 0) u[r] = red[0] * SCALE;
    __syncthreads();
  }
  for (int o = tid; o < Do; o += 256) {
    float s = 0.f;
    for (int r = 0; r < Rr; ++r) s += u[r] * Ball[((size_t)idx * Do + o) * Rr + r];
    out[(size_t)n * Do + o] += s;
  }
}

// ---------------- launcher ----------------
extern "C" void kernel_launch(void* const* d_in, const int* in_sizes, int n_in,
                              void* d_out, int out_size, void* d_ws, size_t ws_size,
                              hipStream_t stream) {
  const float* x    = (const float*)d_in[0];
  const float* W    = (const float*)d_in[1];
  const float* bias = (const float*)d_in[2];
  const float* Aall = (const float*)d_in[3];
  const float* Ball = (const float*)d_in[4];
  const int*   lidx = (const int*)d_in[5];
  float* out = (float*)d_out;

  const size_t XB = (size_t)Nn * Ke * 2;   // 142,606,336
  const size_t WB = (size_t)Do * Ke * 2;   //   8,912,896
  const size_t AB = (size_t)LR * Dk * 2;   //     524,288

  if (ws_size < XB + WB + AB) {            // fallback: no workspace path
    k_fb_base<<<(int)(((size_t)Nn * Do + 255) / 256), 256, 0, stream>>>(x, W, bias, out);
    k_fb_delta<<<Nn, 256, 0, stream>>>(x, Aall, Ball, lidx, out);
    return;
  }

  unsigned short* Xb = (unsigned short*)d_ws;
  unsigned short* Wb = (unsigned short*)((char*)d_ws + XB);
  unsigned short* Ab = (unsigned short*)((char*)d_ws + XB + WB);

  k_build_wb<<<(int)(((size_t)Do * Ke + 255) / 256), 256, 0, stream>>>(W, Ball, Wb);
  k_build_ab<<<(int)(((size_t)LR * Dk + 255) / 256), 256, 0, stream>>>(Aall, Ab);
  k_convert_x<<<(int)(((size_t)Nn * Dk / 8 + 255) / 256), 256, 0, stream>>>(x, Xb);

  // U = mask(scale * (x @ A^T)) -> bf16 columns Xb[:, 2048:2176]
  gemm_wmma<1, Ke, Dk, Dk><<<dim3(Nn / BM, 1), 256, 0, stream>>>(
      Xb, Ab, nullptr, nullptr, Xb, lidx);

  // out = Xb[:, 0:2176] @ Wb^T + bias  (base GEMM + fused LoRA up-projection)
  gemm_wmma<0, Ke, Ke, Ke><<<dim3(Nn / BM, Do / BN), 256, 0, stream>>>(
      Xb, Wb, out, bias, nullptr, nullptr);
}
